// Best_Net_1022202216752
// MI455X (gfx1250) — matrix operand order repaired and checked by reading, not compiled, not run on hardware
//
#include <hip/hip_runtime.h>

typedef __attribute__((ext_vector_type(16))) __bf16 v16bf;
typedef __attribute__((ext_vector_type(8)))  __bf16 v8bf;
typedef __attribute__((ext_vector_type(8)))  float  v8f;
typedef __attribute__((__vector_size__(4 * sizeof(int)))) int v4i_;

constexpr int Hh = 128, Wh = 128, Lq = Hh * Wh;   // 16384
constexpr int Cc = 32, DI = 64, DSn = 16;
constexpr int NCHUNK = 64, CLEN = 256;            // 64 * 256 = 16384

// ---------------------------------------------------------------------------
// CDNA5 async global->LDS staging (guarded: falls back to plain copies)
// ---------------------------------------------------------------------------
#define ASG __attribute__((address_space(1)))
#define ASL __attribute__((address_space(3)))
#if defined(__AMDGCN__) && defined(__has_builtin)
#  if __has_builtin(__builtin_amdgcn_global_load_async_to_lds_b128) && \
      __has_builtin(__builtin_amdgcn_s_wait_asynccnt)
#    define HAVE_ASYNC 1
#  endif
#endif
#ifndef HAVE_ASYNC
#  define HAVE_ASYNC 0
#endif

// stage NFLOAT f32 (multiple of 4) from global into LDS, then barrier
template <int NFLOAT>
__device__ __forceinline__ void stage_w(const float* __restrict__ g,
                                        float* __restrict__ s) {
#if HAVE_ASYNC
  for (int i = threadIdx.x * 4; i < NFLOAT; i += blockDim.x * 4)
    __builtin_amdgcn_global_load_async_to_lds_b128((ASG v4i_*)(g + i),
                                                   (ASL v4i_*)(s + i), 0, 0);
  __builtin_amdgcn_s_wait_asynccnt(0);
#else
  for (int i = threadIdx.x; i < NFLOAT; i += blockDim.x) s[i] = g[i];
#endif
  __syncthreads();
}

// ---------------------------------------------------------------------------
// WMMA helpers (CDNA5 wave32, 16x16x32 bf16 -> f32)
// ---------------------------------------------------------------------------
__device__ __forceinline__ v8f wmma_bf16(v16bf a, v16bf b, v8f c) {
  return __builtin_amdgcn_wmma_f32_16x16x32_bf16(false, a, false, b, (short)0, c,
                                                 false, false);
}

__device__ __forceinline__ v16bf cat16(v8bf lo, v8bf hi) {
  return __builtin_shufflevector(lo, hi, 0, 1, 2, 3, 4, 5, 6, 7, 8, 9, 10, 11,
                                 12, 13, 14, 15);
}

// A fragment 16(M) x 32(K) from row-major bf16 (row stride ld elems, ld%8==0):
// lane layout (ISA 7.12.2): two contiguous 8-elem runs at +8*kh and +16+8*kh.
__device__ __forceinline__ v16bf fragA_rm(const __bf16* base, int ld) {
  int lane = threadIdx.x & 31, m = lane & 15, kh = lane >> 4;
  const __bf16* r = base + m * ld + 8 * kh;
  return cat16(*(const v8bf*)r, *(const v8bf*)(r + 16));
}

// Build NF pre-swizzled B fragments (32 lanes x 16 bf16 each) in LDS from
// f(frag, k, n); B lane layout: lane holds col N=lane&15, lanes 0-15 K 0..7,16..23,
// lanes 16-31 K 8..15,24..31. One barrier at the end.
template <int NF, class F>
__device__ __forceinline__ void swizzleB(__bf16* sBf, F f) {
  for (int e = threadIdx.x; e < NF * 512; e += blockDim.x) {
    int frag = e >> 9, lane = (e >> 4) & 31, j = e & 15;
    int n = lane & 15, k = (j & 7) + 8 * (lane >> 4) + 16 * (j >> 3);
    sBf[e] = (__bf16)f(frag, k, n);
  }
  __syncthreads();
}
// load one pre-swizzled B fragment: a single aligned 32-byte LDS read per lane
__device__ __forceinline__ v16bf ldB(const __bf16* sBf, int frag) {
  return *(const v16bf*)(sBf + (frag * 32 + (threadIdx.x & 31)) * 16);
}

// C/D 16x16 f32: VGPR r holds row M = r + 8*(lane>=16), col N = lane&15.
template <class F>
__device__ __forceinline__ void stC(v8f c, F f) {
  int lane = threadIdx.x & 31, n = lane & 15, mh = lane >> 4;
#pragma unroll
  for (int r = 0; r < 8; ++r) f(r + 8 * mh, n, c[r]);
}

__device__ __forceinline__ int iclamp(int v, int lo, int hi) {
  return v < lo ? lo : (v > hi ? hi : v);
}
__device__ __forceinline__ float siluf(float x) { return x / (1.f + __expf(-x)); }

// ---------------------------------------------------------------------------
// Elementwise / image kernels
// ---------------------------------------------------------------------------
__global__ void k_zero(float* a, int n) {
  int i = blockIdx.x * blockDim.x + threadIdx.x;
  if (i < n) a[i] = 0.f;
}
__global__ void k_add_inplace(float* a, const float* b, int n) {
  int i = blockIdx.x * blockDim.x + threadIdx.x;
  if (i < n) a[i] += b[i];
}

__global__ void k_resize_bilinear(const float* in, float* out, int C, int ih,
                                  int iw, int oh, int ow) {
  int idx = blockIdx.x * blockDim.x + threadIdx.x;
  int total = C * oh * ow;
  if (idx >= total) return;
  int x = idx % ow, y = (idx / ow) % oh, c = idx / (ow * oh);
  float sy = (y + 0.5f) * (float)ih / (float)oh - 0.5f;
  float sx = (x + 0.5f) * (float)iw / (float)ow - 0.5f;
  int y0 = (int)floorf(sy), x0 = (int)floorf(sx);
  float fy = sy - (float)y0, fx = sx - (float)x0;
  int y0c = iclamp(y0, 0, ih - 1), y1c = iclamp(y0 + 1, 0, ih - 1);
  int x0c = iclamp(x0, 0, iw - 1), x1c = iclamp(x0 + 1, 0, iw - 1);
  const float* p = in + c * ih * iw;
  float v = (1.f - fy) * ((1.f - fx) * p[y0c * iw + x0c] + fx * p[y0c * iw + x1c]) +
            fy * ((1.f - fx) * p[y1c * iw + x0c] + fx * p[y1c * iw + x1c]);
  out[idx] = v;
}

__global__ void k_conv3x3(const float* in, const float* w, const float* b,
                          float* out, int Cin, int Cout) {
  int idx = blockIdx.x * blockDim.x + threadIdx.x;
  if (idx >= Cout * Lq) return;
  int x = idx & (Wh - 1), y = (idx >> 7) & (Hh - 1), co = idx / Lq;
  float acc = b[co];
  for (int ci = 0; ci < Cin; ++ci)
    for (int ky = 0; ky < 3; ++ky) {
      int yy = y + ky - 1;
      if ((unsigned)yy >= (unsigned)Hh) continue;
      for (int kx = 0; kx < 3; ++kx) {
        int xx = x + kx - 1;
        if ((unsigned)xx >= (unsigned)Wh) continue;
        acc += w[((co * Cin + ci) * 3 + ky) * 3 + kx] * in[ci * Lq + yy * Wh + xx];
      }
    }
  out[idx] = acc;
}

__global__ void k_blur_down(const float* in, float* out) {
  int idx = blockIdx.x * blockDim.x + threadIdx.x;
  if (idx >= Cc * 64 * 64) return;
  int x = idx & 63, y = (idx >> 6) & 63, c = idx >> 12;
  const float kw[3] = {1.f, 2.f, 1.f};
  float acc = 0.f;
  for (int ky = 0; ky < 3; ++ky) {
    int yy = 2 * y - 1 + ky;
    if ((unsigned)yy >= (unsigned)Hh) continue;
    for (int kx = 0; kx < 3; ++kx) {
      int xx = 2 * x - 1 + kx;
      if ((unsigned)xx >= (unsigned)Wh) continue;
      acc += kw[ky] * kw[kx] * (1.f / 16.f) * in[c * Lq + yy * Wh + xx];
    }
  }
  out[idx] = acc;
}

__device__ __forceinline__ float up2_sample(const float* p, int y, int x) {
  float sy = 0.5f * (float)y - 0.25f, sx = 0.5f * (float)x - 0.25f;
  int y0 = (int)floorf(sy), x0 = (int)floorf(sx);
  float fy = sy - (float)y0, fx = sx - (float)x0;
  int y0c = iclamp(y0, 0, 63), y1c = iclamp(y0 + 1, 0, 63);
  int x0c = iclamp(x0, 0, 63), x1c = iclamp(x0 + 1, 0, 63);
  return (1.f - fy) * ((1.f - fx) * p[y0c * 64 + x0c] + fx * p[y0c * 64 + x1c]) +
         fy * ((1.f - fx) * p[y1c * 64 + x0c] + fx * p[y1c * 64 + x1c]);
}

__global__ void k_cross_rec(const float* m_in, const float* p_in,
                            const float* low_m, const float* low_p, float* m_o,
                            float* p_o) {
  int idx = blockIdx.x * blockDim.x + threadIdx.x;
  if (idx >= Cc * Lq) return;
  int x = idx & (Wh - 1), y = (idx >> 7) & (Hh - 1), c = idx / Lq;
  float um = up2_sample(low_m + c * 4096, y, x);
  float up = up2_sample(low_p + c * 4096, y, x);
  m_o[idx] = um + p_in[idx] - up;
  p_o[idx] = up + m_in[idx] - um;
}

// ---------------------------------------------------------------------------
// HIN conv (32->32, 3x3) implicit GEMM: LDS-staged zero-padded halo tile (bf16)
// + pre-swizzled B fragments in LDS. Block: 64 pixels x 32 outch, 8 waves.
// out = x + bn(lrelu(conv3x3(x)))
// ---------------------------------------------------------------------------
__global__ void k_hin_conv_wmma(const float* __restrict__ x,
                                const float* __restrict__ w,
                                const float* __restrict__ b,
                                const float* __restrict__ g,
                                const float* __restrict__ be,
                                float* __restrict__ out) {
  __shared__ __align__(32) __bf16 sA[3 * 66 * 32];   // halo tile, pixel-major
  __shared__ __align__(32) __bf16 sB[18 * 512];      // 9 taps x 2 ntiles frags
  int p0b = blockIdx.x * 64;
  int ybase = p0b >> 7, xbase = p0b & 127;

  for (int e = threadIdx.x; e < 3 * 66 * 32; e += 256) {
    int ch = e & 31, pix = e >> 5;
    int ix = pix % 66, iy = pix / 66;
    int gy = ybase + iy - 1, gx = xbase + ix - 1;
    float v = 0.f;
    if ((unsigned)gy < (unsigned)Hh && (unsigned)gx < (unsigned)Wh)
      v = x[ch * Lq + gy * Wh + gx];
    sA[pix * 32 + ch] = (__bf16)v;
  }
  swizzleB<18>(sB, [&](int frag, int k, int n) -> float {
    int tap = frag >> 1, nt = frag & 1;
    int dy = tap / 3, dx = tap % 3;
    return w[(((nt * 16 + n) * Cc + k) * 3 + dy) * 3 + dx];
  });

  int wv = threadIdx.x >> 5, lane = threadIdx.x & 31;
  int q = wv >> 1, nt = wv & 1;
  int m = lane & 15, kh = lane >> 4;
  v8f acc = {};
#pragma unroll
  for (int tap = 0; tap < 9; ++tap) {
    int dy = tap / 3, dx = tap % 3;
    const __bf16* ar = sA + (dy * 66 + q * 16 + m + dx) * 32 + 8 * kh;
    v16bf A = cat16(*(const v8bf*)ar, *(const v8bf*)(ar + 16));
    acc = wmma_bf16(A, ldB(sB, tap * 2 + nt), acc);
  }
  int n0 = nt * 16, p0 = p0b + q * 16;
  stC(acc, [&](int mm, int nn, float v) {
    int o = n0 + nn, p = p0 + mm;
    float z = v + b[o];
    z = z > 0.f ? z : 0.2f * z;
    z = z * (g[o] * 0.9999950000374997f) + be[o];  // g/sqrt(1+1e-5)
    out[o * Lq + p] = x[o * Lq + p] + z;
  });
}

// 1x1 conv 32->32, image layout in -> sequence layout out (L x 32)
__global__ void k_conv1x1_seq_wmma(const float* __restrict__ in,
                                   const float* __restrict__ w,
                                   const float* __restrict__ b,
                                   float* __restrict__ out_seq) {
  __shared__ __align__(32) float ws[Cc * Cc];
  __shared__ __align__(32) __bf16 sB[2 * 512];
  __shared__ __align__(32) __bf16 sa[64 * Cc];
  int p0b = blockIdx.x * 64;
  for (int e = threadIdx.x; e < 64 * Cc; e += 256) {
    int ch = e & 31, px = e >> 5;
    sa[px * 32 + ch] = (__bf16)in[ch * Lq + p0b + px];
  }
  stage_w<Cc * Cc>(w, ws);
  swizzleB<2>(sB, [&](int frag, int k, int n) -> float {
    return ws[(frag * 16 + n) * Cc + k];
  });
  int wv = threadIdx.x >> 5;
  int q = wv >> 1, nt = wv & 1, n0 = nt * 16;
  v16bf A = fragA_rm(sa + q * 16 * 32, 32);
  v8f acc = {};
  acc = wmma_bf16(A, ldB(sB, nt), acc);
  int p0 = p0b + q * 16;
  stC(acc, [&](int m, int n, float v) {
    out_seq[(p0 + m) * Cc + (n0 + n)] = v + b[n0 + n];
  });
}

// ---------------------------------------------------------------------------
// Mamba kernels
// ---------------------------------------------------------------------------
__global__ void k_ln_bf16(const float* __restrict__ in, const float* lw,
                          const float* lb, __bf16* __restrict__ out) {
  int l = blockIdx.x * blockDim.x + threadIdx.x;
  if (l >= Lq) return;
  const float* r = in + l * Cc;
  float mu = 0.f;
#pragma unroll
  for (int i = 0; i < Cc; ++i) mu += r[i];
  mu *= (1.f / Cc);
  float v = 0.f;
#pragma unroll
  for (int i = 0; i < Cc; ++i) { float d = r[i] - mu; v += d * d; }
  v *= (1.f / Cc);
  float inv = rsqrtf(v + 1e-5f);
#pragma unroll
  for (int i = 0; i < Cc; ++i)
    out[l * Cc + i] = (__bf16)((r[i] - mu) * inv * lw[i] + lb[i]);
}

// xz = xn @ in_proj.T : (L x 32) x (32 x 128); W async-staged + pre-swizzled,
// A loaded once per wave (2 x b128) and reused across all 8 n-tiles.
__global__ void k_gemm_inproj(const __bf16* __restrict__ xn,
                              const float* __restrict__ W,
                              float* __restrict__ xz) {
  __shared__ __align__(32) float ws[128 * Cc];
  __shared__ __align__(32) __bf16 sB[8 * 512];
  stage_w<128 * Cc>(W, ws);
  swizzleB<8>(sB, [&](int frag, int k, int n) -> float {
    return ws[(frag * 16 + n) * Cc + k];
  });
  int wv = threadIdx.x >> 5;
  int p0 = (blockIdx.x * 8 + wv) * 16;
  __builtin_prefetch(xn + p0 * Cc, 0, 0);
  v16bf A = fragA_rm(xn + p0 * Cc, Cc);
#pragma unroll
  for (int nt = 0; nt < 8; ++nt) {
    v8f acc = {};
    acc = wmma_bf16(A, ldB(sB, nt), acc);
    stC(acc, [&](int m, int n, float v) { xz[(p0 + m) * 128 + nt * 16 + n] = v; });
  }
}

// depthwise causal conv (window 4, left pad 3) + silu; f32 + bf16 copies
__global__ void k_dwconv_silu(const float* __restrict__ xz,
                              const float* __restrict__ cw,
                              const float* __restrict__ cb,
                              float* __restrict__ xs_act,
                              __bf16* __restrict__ xs_bf) {
  int idx = blockIdx.x * blockDim.x + threadIdx.x;
  if (idx >= Lq * DI) return;
  int d = idx & (DI - 1), l = idx >> 6;
  float acc = cb[d];
#pragma unroll
  for (int j = 0; j < 4; ++j) {
    int ll = l - 3 + j;
    if (ll >= 0) acc += cw[d * 4 + j] * xz[ll * 128 + d];
  }
  float s = siluf(acc);
  xs_act[idx] = s;
  xs_bf[idx] = (__bf16)s;
}

// x_dbl = xs @ x_proj.T : (L x 64) x (64 x 34), N padded to 48
__global__ void k_gemm_xproj(const __bf16* __restrict__ xs,
                             const float* __restrict__ W,
                             float* __restrict__ xdbl) {
  __shared__ __align__(32) float ws[34 * DI + 2];  // 2176 floats (mult of 4)
  __shared__ __align__(32) __bf16 sB[6 * 512];     // 3 ntiles x 2 k-steps
  stage_w<34 * DI>(W, ws);
  swizzleB<6>(sB, [&](int frag, int k, int n) -> float {
    int row = (frag >> 1) * 16 + n, kk = (frag & 1) * 32;
    return row < 34 ? ws[row * DI + kk + k] : 0.f;
  });
  int wv = threadIdx.x >> 5;
  int p0 = (blockIdx.x * 8 + wv) * 16;
  __builtin_prefetch(xs + p0 * DI, 0, 0);
  v16bf A0 = fragA_rm(xs + p0 * DI, DI);
  v16bf A1 = fragA_rm(xs + p0 * DI + 32, DI);
#pragma unroll
  for (int nt = 0; nt < 3; ++nt) {
    v8f acc = {};
    acc = wmma_bf16(A0, ldB(sB, nt * 2 + 0), acc);
    acc = wmma_bf16(A1, ldB(sB, nt * 2 + 1), acc);
    stC(acc, [&](int m, int n, float v) { xdbl[(p0 + m) * 48 + nt * 16 + n] = v; });
  }
}

__global__ void k_delta(const float* __restrict__ xdbl, const float* dtw,
                        const float* dtb, float* __restrict__ delta) {
  int idx = blockIdx.x * blockDim.x + threadIdx.x;
  if (idx >= Lq * DI) return;
  int d = idx & (DI - 1), l = idx >> 6;
  float t = xdbl[l * 48] * dtw[d * 2] + xdbl[l * 48 + 1] * dtw[d * 2 + 1] + dtb[d];
  delta[idx] = t > 20.f ? t : log1pf(__expf(t));
}

// scan pass 1: per (chunk, d, n) chunk product P and local end-state S
__global__ void k_scan1(const float* __restrict__ delta,
                        const float* __restrict__ xs_act,
                        const float* __restrict__ xdbl,
                        const float* __restrict__ A_log, float* __restrict__ P,
                        float* __restrict__ S) {
  int tid = blockIdx.x * blockDim.x + threadIdx.x;  // 64 * 1024
  int c = tid >> 10, s = tid & 1023, d = s >> 4, n = s & 15;
  float Ac = -__expf(A_log[d * DSn + n]);
  float h = 0.f, p = 1.f;
  int l0 = c * CLEN;
  for (int i = 0; i < CLEN; ++i) {
    int l = l0 + i;
    float dl = delta[l * DI + d];
    float a = __expf(dl * Ac);
    h = a * h + dl * xs_act[l * DI + d] * xdbl[l * 48 + 2 + n];
    p *= a;
  }
  P[tid] = p;
  S[tid] = h;
}

// scan pass 2: serial prefix across the 64 chunks per state channel
__global__ void k_scan2(const float* __restrict__ P, const float* __restrict__ S,
                        float* __restrict__ H0) {
  int s = blockIdx.x * blockDim.x + threadIdx.x;
  if (s >= 1024) return;
  float h = 0.f;
  for (int c = 0; c < NCHUNK; ++c) {
    H0[c * 1024 + s] = h;
    h = P[c * 1024 + s] * h + S[c * 1024 + s];
  }
}

// scan pass 3: replay with correct h0, reduce over n (16 lanes), gate, write y
__global__ void k_scan3(const float* __restrict__ delta,
                        const float* __restrict__ xs_act,
                        const float* __restrict__ xdbl,
                        const float* __restrict__ A_log,
                        const float* __restrict__ H0, const float* __restrict__ Dp,
                        const float* __restrict__ xz, __bf16* __restrict__ ygbf) {
  int c = blockIdx.x >> 2, dg = blockIdx.x & 3;
  int t = threadIdx.x;
  int n = t & 15, d = dg * 16 + (t >> 4);
  float Ac = -__expf(A_log[d * DSn + n]);
  float h = H0[c * 1024 + d * DSn + n];
  int l0 = c * CLEN;
  for (int i = 0; i < CLEN; ++i) {
    int l = l0 + i;
    float dl = delta[l * DI + d];
    float a = __expf(dl * Ac);
    float u = xs_act[l * DI + d];
    h = a * h + dl * u * xdbl[l * 48 + 2 + n];
    float yc = h * xdbl[l * 48 + 18 + n];
    yc += __shfl_xor(yc, 1, 32);
    yc += __shfl_xor(yc, 2, 32);
    yc += __shfl_xor(yc, 4, 32);
    yc += __shfl_xor(yc, 8, 32);
    if (n == 0) {
      float yv = yc + u * Dp[d];
      float zv = xz[l * 128 + DI + d];
      yv *= siluf(zv);
      ygbf[l * DI + d] = (__bf16)yv;
    }
  }
}

// out = y @ out_proj.T : (L x 64) x (64 x 32)
__global__ void k_gemm_outproj(const __bf16* __restrict__ yg,
                               const float* __restrict__ W,
                               float* __restrict__ out_seq) {
  __shared__ __align__(32) float ws[Cc * DI];
  __shared__ __align__(32) __bf16 sB[4 * 512];  // 2 ntiles x 2 k-steps
  stage_w<Cc * DI>(W, ws);
  swizzleB<4>(sB, [&](int frag, int k, int n) -> float {
    return ws[((frag >> 1) * 16 + n) * DI + (frag & 1) * 32 + k];
  });
  int wv = threadIdx.x >> 5;
  int p0 = (blockIdx.x * 8 + wv) * 16;
  __builtin_prefetch(yg + p0 * DI, 0, 0);
  v16bf A0 = fragA_rm(yg + p0 * DI, DI);
  v16bf A1 = fragA_rm(yg + p0 * DI + 32, DI);
#pragma unroll
  for (int nt = 0; nt < 2; ++nt) {
    v8f acc = {};
    acc = wmma_bf16(A0, ldB(sB, nt * 2 + 0), acc);
    acc = wmma_bf16(A1, ldB(sB, nt * 2 + 1), acc);
    stC(acc, [&](int m, int n, float v) { out_seq[(p0 + m) * Cc + nt * 16 + n] = v; });
  }
}

// final: out = conv3x3(m_img,32->4) + conv3x3(p_img,32->1) + m_up (seq inputs)
__global__ void k_final(const float* __restrict__ mseq, const float* __restrict__ pseq,
                        const float* mw, const float* mb, const float* pw,
                        const float* pb, const float* __restrict__ mup,
                        float* __restrict__ out) {
  int idx = blockIdx.x * blockDim.x + threadIdx.x;
  if (idx >= 4 * Lq) return;
  int x = idx & (Wh - 1), y = (idx >> 7) & (Hh - 1), cch = idx / Lq;
  float am = mb[cch], ap = pb[0];
  for (int ky = 0; ky < 3; ++ky) {
    int yy = y + ky - 1;
    if ((unsigned)yy >= (unsigned)Hh) continue;
    for (int kx = 0; kx < 3; ++kx) {
      int xx = x + kx - 1;
      if ((unsigned)xx >= (unsigned)Wh) continue;
      int p = yy * Wh + xx;
      for (int ci = 0; ci < Cc; ++ci) {
        am += mw[((cch * Cc + ci) * 3 + ky) * 3 + kx] * mseq[p * Cc + ci];
        ap += pw[(ci * 3 + ky) * 3 + kx] * pseq[p * Cc + ci];
      }
    }
  }
  out[idx] = am + ap + mup[idx];
}

// ---------------------------------------------------------------------------
// Host orchestration
// ---------------------------------------------------------------------------
extern "C" void kernel_launch(void* const* d_in, const int* in_sizes, int n_in,
                              void* d_out, int out_size, void* d_ws, size_t ws_size,
                              hipStream_t stream) {
  (void)in_sizes; (void)n_in; (void)out_size; (void)ws_size;
  int ci = 0;
  auto nxt = [&]() -> const float* { return (const float*)d_in[ci++]; };

  const float* ms = nxt();
  const float* pan = nxt();
  const float* m_in_w = nxt(); const float* m_in_b = nxt();
  const float* p_in_w = nxt(); const float* p_in_b = nxt();

  struct Hin { const float *w1, *b1, *g1, *be1, *w2, *b2, *g2, *be2; };
  Hin menc[3], penc[3];
  for (int i = 0; i < 3; ++i) {
    menc[i].w1 = nxt(); menc[i].b1 = nxt(); menc[i].g1 = nxt(); menc[i].be1 = nxt();
    menc[i].w2 = nxt(); menc[i].b2 = nxt(); menc[i].g2 = nxt(); menc[i].be2 = nxt();
  }
  for (int i = 0; i < 3; ++i) {
    penc[i].w1 = nxt(); penc[i].b1 = nxt(); penc[i].g1 = nxt(); penc[i].be1 = nxt();
    penc[i].w2 = nxt(); penc[i].b2 = nxt(); penc[i].g2 = nxt(); penc[i].be2 = nxt();
  }
  const float* m_part_w = nxt(); const float* m_part_b = nxt();
  const float* p_part_w = nxt(); const float* p_part_b = nxt();

  struct MP { const float *ln_w, *ln_b, *in_proj, *conv_w, *conv_b, *x_proj,
                          *dt_w, *dt_b, *A_log, *D, *out_proj; };
  MP inv[8], mmb[8], pmb[8];
  auto read_mp = [&](MP& p) {
    p.ln_w = nxt(); p.ln_b = nxt(); p.in_proj = nxt(); p.conv_w = nxt();
    p.conv_b = nxt(); p.x_proj = nxt(); p.dt_w = nxt(); p.dt_b = nxt();
    p.A_log = nxt(); p.D = nxt(); p.out_proj = nxt();
  };
  for (int i = 0; i < 8; ++i) read_mp(inv[i]);
  for (int i = 0; i < 8; ++i) read_mp(mmb[i]);
  for (int i = 0; i < 8; ++i) read_mp(pmb[i]);
  const float* m_out_w = nxt(); const float* m_out_b = nxt();
  const float* p_out_w = nxt(); const float* p_out_b = nxt();

  // --- workspace bump allocator (256B aligned) ---
  char* wsb = (char*)d_ws;
  size_t off = 0;
  auto alloc = [&](size_t bytes) -> void* {
    void* p = wsb + off;
    off = (off + bytes + 255) & ~(size_t)255;
    return p;
  };
  float* m_up  = (float*)alloc(4 * Lq * 4);
  float* m_inA = (float*)alloc(Cc * Lq * 4);
  float* p_inA = (float*)alloc(Cc * Lq * 4);
  float* low_m = (float*)alloc(Cc * 4096 * 4);
  float* low_p = (float*)alloc(Cc * 4096 * 4);
  float* m_o   = (float*)alloc(Cc * Lq * 4);
  float* p_o   = (float*)alloc(Cc * Lq * 4);
  float* t0    = (float*)alloc(Cc * Lq * 4);
  float* seq0  = (float*)alloc(Lq * Cc * 4);
  float* seq1  = (float*)alloc(Lq * Cc * 4);
  float* seq2  = (float*)alloc(Lq * Cc * 4);
  float* seq3  = (float*)alloc(Lq * Cc * 4);
  float* seq4  = (float*)alloc(Lq * Cc * 4);
  __bf16* xn_bf  = (__bf16*)alloc(Lq * Cc * 2);
  float* xz      = (float*)alloc((size_t)Lq * 128 * 4);
  float* xs_act  = (float*)alloc((size_t)Lq * DI * 4);
  __bf16* xs_bf  = (__bf16*)alloc((size_t)Lq * DI * 2);
  float* xdbl    = (float*)alloc((size_t)Lq * 48 * 4);
  float* dlt     = (float*)alloc((size_t)Lq * DI * 4);
  float* scanP   = (float*)alloc(NCHUNK * 1024 * 4);
  float* scanS   = (float*)alloc(NCHUNK * 1024 * 4);
  float* scanH0  = (float*)alloc(NCHUNK * 1024 * 4);
  __bf16* yg_bf  = (__bf16*)alloc((size_t)Lq * DI * 2);

  const dim3 B256(256);

  // ---- stem ----
  k_resize_bilinear<<<dim3(256), B256, 0, stream>>>(ms, m_up, 4, 32, 32, Hh, Wh);
  k_conv3x3<<<dim3(2048), B256, 0, stream>>>(m_up, m_in_w, m_in_b, m_inA, 4, Cc);
  k_conv3x3<<<dim3(2048), B256, 0, stream>>>(pan, p_in_w, p_in_b, p_inA, 1, Cc);
  k_blur_down<<<dim3(512), B256, 0, stream>>>(m_inA, low_m);
  k_blur_down<<<dim3(512), B256, 0, stream>>>(p_inA, low_p);
  k_cross_rec<<<dim3(2048), B256, 0, stream>>>(m_inA, p_inA, low_m, low_p, m_o, p_o);

  // ---- HIN encoders (WMMA implicit GEMM, LDS-staged) ----
  for (int i = 0; i < 3; ++i) {
    k_hin_conv_wmma<<<dim3(256), B256, 0, stream>>>(m_o, menc[i].w1, menc[i].b1,
                                                    menc[i].g1, menc[i].be1, t0);
    k_hin_conv_wmma<<<dim3(256), B256, 0, stream>>>(t0, menc[i].w2, menc[i].b2,
                                                    menc[i].g2, menc[i].be2, m_o);
  }
  for (int i = 0; i < 3; ++i) {
    k_hin_conv_wmma<<<dim3(256), B256, 0, stream>>>(p_o, penc[i].w1, penc[i].b1,
                                                    penc[i].g1, penc[i].be1, t0);
    k_hin_conv_wmma<<<dim3(256), B256, 0, stream>>>(t0, penc[i].w2, penc[i].b2,
                                                    penc[i].g2, penc[i].be2, p_o);
  }

  // ---- 1x1 projections to sequence layout ----
  k_conv1x1_seq_wmma<<<dim3(256), B256, 0, stream>>>(m_o, m_part_w, m_part_b, seq0);
  k_conv1x1_seq_wmma<<<dim3(256), B256, 0, stream>>>(p_o, p_part_w, p_part_b, seq1);

  // ---- mamba block driver ----
  auto run_mamba = [&](const float* xin_seq, float* out_seq, const MP& p) {
    k_ln_bf16<<<dim3(64), B256, 0, stream>>>(xin_seq, p.ln_w, p.ln_b, xn_bf);
    k_gemm_inproj<<<dim3(128), B256, 0, stream>>>(xn_bf, p.in_proj, xz);
    k_dwconv_silu<<<dim3(4096), B256, 0, stream>>>(xz, p.conv_w, p.conv_b, xs_act, xs_bf);
    k_gemm_xproj<<<dim3(128), B256, 0, stream>>>(xs_bf, p.x_proj, xdbl);
    k_delta<<<dim3(4096), B256, 0, stream>>>(xdbl, p.dt_w, p.dt_b, dlt);
    k_scan1<<<dim3(256), B256, 0, stream>>>(dlt, xs_act, xdbl, p.A_log, scanP, scanS);
    k_scan2<<<dim3(4), B256, 0, stream>>>(scanP, scanS, scanH0);
    k_scan3<<<dim3(256), B256, 0, stream>>>(dlt, xs_act, xdbl, p.A_log, scanH0,
                                            p.D, xz, yg_bf);
    k_gemm_outproj<<<dim3(128), B256, 0, stream>>>(yg_bf, p.out_proj, out_seq);
  };

  // ---- inverted mamba stack: res += x; xn = mamba(ln(res)); x,res = res,xn ----
  float *px = seq0, *pres = seq1, *pfree = seq2;
  for (int i = 0; i < 8; ++i) {
    k_add_inplace<<<dim3(2048), B256, 0, stream>>>(pres, px, Lq * Cc);
    run_mamba(pres, pfree, inv[i]);
    float* oldx = px;
    px = pres; pres = pfree; pfree = oldx;
  }
  float* m_seq = px;
  float* p_seq = pres;

  // ---- run(m_seq, m_mamba) ----
  k_zero<<<dim3(2048), B256, 0, stream>>>(seq3, Lq * Cc);
  float* pxx = m_seq; float* pf = seq4;
  for (int i = 0; i < 8; ++i) {
    k_add_inplace<<<dim3(2048), B256, 0, stream>>>(seq3, pxx, Lq * Cc);
    run_mamba(seq3, pf, mmb[i]);
    float* tsw = pxx; pxx = pf; pf = tsw;
  }
  float* m_final = pxx;

  // ---- run(p_seq, p_mamba) ----
  k_zero<<<dim3(2048), B256, 0, stream>>>(seq3, Lq * Cc);
  float* pxx2 = p_seq; float* pf2 = pfree;
  for (int i = 0; i < 8; ++i) {
    k_add_inplace<<<dim3(2048), B256, 0, stream>>>(seq3, pxx2, Lq * Cc);
    run_mamba(seq3, pf2, pmb[i]);
    float* tsw = pxx2; pxx2 = pf2; pf2 = tsw;
  }
  float* p_final = pxx2;

  // ---- output head ----
  k_final<<<dim3(256), B256, 0, stream>>>(m_final, p_final, m_out_w, m_out_b,
                                          p_out_w, p_out_b, m_up, (float*)d_out);
}
